// MSA_Block_56770877718955
// MI455X (gfx1250) — matrix operand order repaired and checked by reading, not compiled, not run on hardware
//
#include <hip/hip_runtime.h>

// ---------------------------------------------------------------------------
// MSA block for gfx1250 (MI455X), bf16 WMMA (v_wmma_f32_16x16x32_bf16) path,
// with async global->LDS staging (ASYNCcnt) in the flash-attention kernel.
// d_ws layout (uint16_t bf16 elements). Requires ws_size >= 92,274,688 bytes.
// ---------------------------------------------------------------------------

#define BATCH   4
#define HEADS   16
#define NSEQ    2048
#define DH      64
#define DMODEL  1024
#define DQKV    3072
#define MROWS   (BATCH * NSEQ)          // 8192

// Workspace offsets (in uint16_t elements)
#define OFF_XB    ((size_t)0)                    // x bf16         [8192][1024]
#define OFF_WQKV  ((size_t)8388608)              // w_qkv bf16     [3072][1024]
#define OFF_WOUT  ((size_t)11534336)             // w_out bf16     [1024][1024]
#define OFF_Q     ((size_t)12582912)             // Q bf16         [b][h][n][d]
#define OFF_K     ((size_t)20971520)             // K bf16         [b][h][n][d]
#define OFF_V     ((size_t)29360128)             // V bf16 (transp)[b][h][d][n]
#define OFF_AO    ((size_t)37748736)             // attn out bf16  [8192][1024]

typedef __attribute__((ext_vector_type(16))) __bf16 v16bf;
typedef __attribute__((ext_vector_type(8)))  float  v8f;

union BFrag { v16bf v; uint4 q[2]; };

static __device__ __forceinline__ unsigned short f2bf(float f) {
    unsigned int u = __float_as_uint(f);
    return (unsigned short)((u + 0x7fffu + ((u >> 16) & 1u)) >> 16);   // RNE
}

// A-fragment style: two separate 16B chunks
static __device__ __forceinline__ v16bf load_frag2(const unsigned short* p0,
                                                   const unsigned short* p1) {
    BFrag f;
    f.q[0] = *(const uint4*)p0;
    f.q[1] = *(const uint4*)p1;
    return f.v;
}
// B-fragment style: one contiguous 32B chunk
static __device__ __forceinline__ v16bf load_frag32(const unsigned short* p) {
    BFrag f;
    f.q[0] = *(const uint4*)p;
    f.q[1] = *(const uint4*)(p + 8);
    return f.v;
}

static __device__ __forceinline__ v8f wmma_bf16(v16bf a, v16bf b, v8f c) {
    return __builtin_amdgcn_wmma_f32_16x16x32_bf16(
        /*neg_a=*/false, a, /*neg_b=*/false, b,
        /*c_mod=*/(short)0, c, /*reuse_a=*/false, /*reuse_b=*/false);
}

// --- gfx1250 async global->LDS copy (16B per lane), tracked by ASYNCcnt ----
static __device__ __forceinline__ void async_copy16(unsigned lds_addr,
                                                    const void* gaddr) {
    asm volatile("global_load_async_to_lds_b128 %0, %1, off"
                 :: "v"(lds_addr), "v"(gaddr)
                 : "memory");
}
static __device__ __forceinline__ void wait_async0() {
    asm volatile("s_wait_asynccnt 0x0" ::: "memory");
}

// ------------------------------- convert -----------------------------------
__global__ void cvt_f32_bf16(const float* __restrict__ in,
                             unsigned short* __restrict__ out, int n) {
    int i = (blockIdx.x * blockDim.x + threadIdx.x) * 4;
    if (i + 3 < n) {
        float4 f = *(const float4*)(in + i);
        uint2 u;
        u.x = (unsigned)f2bf(f.x) | ((unsigned)f2bf(f.y) << 16);
        u.y = (unsigned)f2bf(f.z) | ((unsigned)f2bf(f.w) << 16);
        *(uint2*)(out + i) = u;
    }
}

// ------------------------------ QKV GEMM -----------------------------------
// C[8192][3072] = Xb[8192][1024] @ Wqkv[3072][1024]^T ; scatter into Q/K/Vt.
__global__ __launch_bounds__(128) void qkv_gemm(unsigned short* __restrict__ ws) {
    const unsigned short* X = ws + OFF_XB;
    const unsigned short* W = ws + OFF_WQKV;

    const int lane = threadIdx.x & 31;
    const int wv   = threadIdx.x >> 5;
    const int half = lane >> 4;
    const int c    = lane & 15;
    const int m0   = blockIdx.x * 128 + (wv >> 1) * 64;
    const int n0   = blockIdx.y * 128 + (wv & 1) * 64;

    v8f zero = {};
    v8f acc[4][4];
#pragma unroll
    for (int i = 0; i < 4; ++i)
#pragma unroll
        for (int j = 0; j < 4; ++j) acc[i][j] = zero;

    for (int k0 = 0; k0 < DMODEL; k0 += 32) {
        v16bf a[4], b[4];
#pragma unroll
        for (int i = 0; i < 4; ++i) {
            const unsigned short* r = X + (size_t)(m0 + 16 * i + c) * DMODEL + k0;
            a[i] = load_frag2(r + half * 16, r + half * 16 + 8);
        }
#pragma unroll
        for (int j = 0; j < 4; ++j) {
            const unsigned short* r =
                W + (size_t)(n0 + 16 * j + c) * DMODEL + k0 + half * 16;
            b[j] = load_frag32(r);
        }
#pragma unroll
        for (int i = 0; i < 4; ++i)
#pragma unroll
            for (int j = 0; j < 4; ++j)
                acc[i][j] = wmma_bf16(a[i], b[j], acc[i][j]);
    }

    // Epilogue: scatter into Q [b,h,n,d], K [b,h,n,d], V transposed [b,h,d,n]
#pragma unroll
    for (int i = 0; i < 4; ++i) {
#pragma unroll
        for (int j = 0; j < 4; ++j) {
            const int col   = n0 + 16 * j + c;
            const int which = col >> 10;          // 0=q 1=k 2=v
            const int inner = col & 1023;
            const int head  = inner >> 6;
            const int dim   = inner & 63;
#pragma unroll
            for (int r = 0; r < 8; ++r) {
                const int row   = m0 + 16 * i + r + 8 * half;
                const int batch = row >> 11;
                const int tok   = row & 2047;
                const unsigned short b16 = f2bf(acc[i][j][r]);
                const size_t bh = (size_t)batch * HEADS + head;
                if (which == 0)
                    ws[OFF_Q + ((bh * NSEQ + tok) * DH + dim)] = b16;
                else if (which == 1)
                    ws[OFF_K + ((bh * NSEQ + tok) * DH + dim)] = b16;
                else
                    ws[OFF_V + ((bh * DH + dim) * NSEQ + tok)] = b16;
            }
        }
    }
}

// --------------------------- flash attention -------------------------------
// Block = 4 waves on ONE (batch, head) with 4 consecutive 16-query tiles.
// K/V 32-key tiles are staged in LDS via async global->LDS copies (double
// buffered, ASYNCcnt), shared by all 4 waves; WMMA fragments read from LDS.
__global__ __launch_bounds__(128) void attn_flash(unsigned short* __restrict__ ws) {
    __shared__ __align__(16) unsigned short Kb[2][32 * 64];   // [key][dim]  8KB
    __shared__ __align__(16) unsigned short Vb[2][64 * 32];   // [dim][seq]  8KB
    __shared__ __align__(16) unsigned short Pb[4][16 * 32];   // P / wave    4KB

    const int t    = threadIdx.x;
    const int lane = t & 31;
    const int wv   = t >> 5;
    const int half = lane >> 4;
    const int c    = lane & 15;

    // blockIdx.x in [0,2048): 32 q-groups x 16 heads x 4 batches
    const int qg    = blockIdx.x & 31;
    const int head  = (blockIdx.x >> 5) & 15;
    const int batch = blockIdx.x >> 9;
    const int bh    = batch * HEADS + head;

    const unsigned short* Q  = ws + OFF_Q + (size_t)bh * NSEQ * DH;
    const unsigned short* Km = ws + OFF_K + (size_t)bh * NSEQ * DH;
    const unsigned short* Vt = ws + OFF_V + (size_t)bh * DH * NSEQ;
    unsigned short* P = Pb[wv];

    const int q0 = qg * 64 + wv * 16;

    // Q A-fragments (16x64 -> two 16x32 fragments), direct from global
    v16bf qa[2];
#pragma unroll
    for (int kk = 0; kk < 2; ++kk) {
        const unsigned short* base = Q + (size_t)(q0 + c) * DH + kk * 32;
        qa[kk] = load_frag2(base + half * 16, base + half * 16 + 8);
    }

    v8f zero = {};
    v8f o[4];
#pragma unroll
    for (int tt = 0; tt < 4; ++tt) o[tt] = zero;
    float mr[8], lr[8];
#pragma unroll
    for (int r = 0; r < 8; ++r) { mr[r] = -3.0e38f; lr[r] = 0.0f; }

    const float sc = 0.125f;   // 1/sqrt(64)

    // --- async staging of one 32-key K tile + V tile (4 x 16B per thread) ---
    const int kkey = t >> 3;            // 0..15 (key row within half-tile)
    const int kd8  = (t & 7) * 8;       // dim chunk
    const int vd   = t >> 2;            // 0..31 (dim row within half-tile)
    const int vs8  = (t & 3) * 8;       // seq chunk
    auto issueKV = [&](int buf, int kb) {
        async_copy16((unsigned)(size_t)&Kb[buf][kkey * 64 + kd8],
                     Km + (size_t)(kb + kkey) * DH + kd8);
        async_copy16((unsigned)(size_t)&Kb[buf][(16 + kkey) * 64 + kd8],
                     Km + (size_t)(kb + 16 + kkey) * DH + kd8);
        async_copy16((unsigned)(size_t)&Vb[buf][vd * 32 + vs8],
                     Vt + (size_t)vd * NSEQ + kb + vs8);
        async_copy16((unsigned)(size_t)&Vb[buf][(vd + 32) * 32 + vs8],
                     Vt + (size_t)(vd + 32) * NSEQ + kb + vs8);
    };

    int curb = 0;
    issueKV(0, 0);

    for (int kb = 0; kb < NSEQ; kb += 32) {
        wait_async0();        // own async copies into curb complete
        __syncthreads();      // all waves' copies visible; prev compute done
        if (kb + 32 < NSEQ) issueKV(curb ^ 1, kb + 32);

        // ---- S = Q @ K^T for 16x32 key block (two 16x16 n-tiles) ----
        v8f s0 = zero, s1 = zero;
#pragma unroll
        for (int kk = 0; kk < 2; ++kk) {
            v16bf kf0 = load_frag32(&Kb[curb][(c)      * 64 + kk * 32 + half * 16]);
            v16bf kf1 = load_frag32(&Kb[curb][(16 + c) * 64 + kk * 32 + half * 16]);
            s0 = wmma_bf16(qa[kk], kf0, s0);
            s1 = wmma_bf16(qa[kk], kf1, s1);
        }

        // ---- online softmax (rows live in half-wave lane groups) ----
        float corr[8];
#pragma unroll
        for (int r = 0; r < 8; ++r) {
            float v0 = s0[r] * sc, v1 = s1[r] * sc;
            float mx = fmaxf(v0, v1);
            mx = fmaxf(mx, __shfl_xor(mx, 1, 32));
            mx = fmaxf(mx, __shfl_xor(mx, 2, 32));
            mx = fmaxf(mx, __shfl_xor(mx, 4, 32));
            mx = fmaxf(mx, __shfl_xor(mx, 8, 32));
            float mn = fmaxf(mr[r], mx);
            float p0 = __expf(v0 - mn);
            float p1 = __expf(v1 - mn);
            float cr = __expf(mr[r] - mn);
            float sm = p0 + p1;
            sm += __shfl_xor(sm, 1, 32);
            sm += __shfl_xor(sm, 2, 32);
            sm += __shfl_xor(sm, 4, 32);
            sm += __shfl_xor(sm, 8, 32);
            lr[r] = lr[r] * cr + sm;
            mr[r] = mn;
            corr[r] = cr;
            const int row = r + 8 * half;            // C/D-layout row
            P[row * 32 + c]      = f2bf(p0);
            P[row * 32 + 16 + c] = f2bf(p1);
        }

        // rescale O accumulators (same fragment layout as corr[])
#pragma unroll
        for (int tt = 0; tt < 4; ++tt)
#pragma unroll
            for (int r = 0; r < 8; ++r) o[tt][r] *= corr[r];

        // ---- O += P (16x32) @ V (32x64): P via LDS transpose to A-layout ----
        const unsigned short* pr = P + (size_t)c * 32;
        v16bf pf = load_frag2(pr + half * 16, pr + half * 16 + 8);
#pragma unroll
        for (int tt = 0; tt < 4; ++tt) {
            v16bf vf = load_frag32(&Vb[curb][(tt * 16 + c) * 32 + half * 16]);
            o[tt] = wmma_bf16(pf, vf, o[tt]);
        }

        curb ^= 1;
    }

    // ---- finalize: O /= l, store to attn-out [8192][1024] (bf16) ----
    float inv[8];
#pragma unroll
    for (int r = 0; r < 8; ++r) inv[r] = 1.0f / lr[r];
#pragma unroll
    for (int tt = 0; tt < 4; ++tt)
#pragma unroll
        for (int r = 0; r < 8; ++r) {
            const int row = q0 + r + 8 * half;
            ws[OFF_AO + ((size_t)batch * NSEQ + row) * DMODEL
                      + head * DH + tt * 16 + c] = f2bf(o[tt][r] * inv[r]);
        }
}

// --------------------------- output projection -----------------------------
// out[8192][1024] (f32) = AO[8192][1024] @ Wout[1024][1024]^T + bias
__global__ __launch_bounds__(128) void out_gemm(const unsigned short* __restrict__ ws,
                                                const float* __restrict__ bias,
                                                float* __restrict__ out) {
    const unsigned short* A = ws + OFF_AO;
    const unsigned short* W = ws + OFF_WOUT;

    const int lane = threadIdx.x & 31;
    const int wv   = threadIdx.x >> 5;
    const int half = lane >> 4;
    const int c    = lane & 15;
    const int m0   = blockIdx.x * 128 + (wv >> 1) * 64;
    const int n0   = blockIdx.y * 128 + (wv & 1) * 64;

    v8f zero = {};
    v8f acc[4][4];
#pragma unroll
    for (int i = 0; i < 4; ++i)
#pragma unroll
        for (int j = 0; j < 4; ++j) acc[i][j] = zero;

    for (int k0 = 0; k0 < DMODEL; k0 += 32) {
        v16bf a[4], b[4];
#pragma unroll
        for (int i = 0; i < 4; ++i) {
            const unsigned short* r = A + (size_t)(m0 + 16 * i + c) * DMODEL + k0;
            a[i] = load_frag2(r + half * 16, r + half * 16 + 8);
        }
#pragma unroll
        for (int j = 0; j < 4; ++j) {
            const unsigned short* r =
                W + (size_t)(n0 + 16 * j + c) * DMODEL + k0 + half * 16;
            b[j] = load_frag32(r);
        }
#pragma unroll
        for (int i = 0; i < 4; ++i)
#pragma unroll
            for (int j = 0; j < 4; ++j)
                acc[i][j] = wmma_bf16(a[i], b[j], acc[i][j]);
    }

#pragma unroll
    for (int i = 0; i < 4; ++i)
#pragma unroll
        for (int j = 0; j < 4; ++j) {
            const int col = n0 + 16 * j + c;
            const float bb = bias[col];
#pragma unroll
            for (int r = 0; r < 8; ++r) {
                const int row = m0 + 16 * i + r + 8 * half;
                out[(size_t)row * DMODEL + col] = acc[i][j][r] + bb;
            }
        }
}

// ------------------------------- launcher ----------------------------------
extern "C" void kernel_launch(void* const* d_in, const int* in_sizes, int n_in,
                              void* d_out, int out_size, void* d_ws, size_t ws_size,
                              hipStream_t stream) {
    const float* x      = (const float*)d_in[0];
    const float* w_qkv  = (const float*)d_in[1];
    const float* w_out  = (const float*)d_in[2];
    const float* b_out  = (const float*)d_in[3];
    unsigned short* ws16 = (unsigned short*)d_ws;
    float* out = (float*)d_out;

    // fp32 -> bf16 staging
    {
        const int n1 = MROWS * DMODEL;      // 8,388,608
        const int n2 = DQKV * DMODEL;       // 3,145,728
        const int n3 = DMODEL * DMODEL;     // 1,048,576
        cvt_f32_bf16<<<n1 / 1024, 256, 0, stream>>>(x,     ws16 + OFF_XB,   n1);
        cvt_f32_bf16<<<n2 / 1024, 256, 0, stream>>>(w_qkv, ws16 + OFF_WQKV, n2);
        cvt_f32_bf16<<<n3 / 1024, 256, 0, stream>>>(w_out, ws16 + OFF_WOUT, n3);
    }

    // QKV projection: [8192,1024] x [1024,3072]
    qkv_gemm<<<dim3(MROWS / 128, DQKV / 128), 128, 0, stream>>>(ws16);

    // Flash attention: 2048 blocks x 4 waves (one (b,h) per block)
    attn_flash<<<2048, 128, 0, stream>>>(ws16);

    // Output projection + bias: [8192,1024] x [1024,1024]
    out_gemm<<<dim3(MROWS / 128, DMODEL / 128), 128, 0, stream>>>(ws16, b_out, out);
}
